// DyNFM_70257075028544
// MI455X (gfx1250) — compile-verified, matrix operand order checked by reading, not run on hardware
//
#include <hip/hip_runtime.h>
#include <hip/hip_bf16.h>

// ---------------------------------------------------------------------------
// Types
// ---------------------------------------------------------------------------
typedef __bf16 bf16;
typedef __attribute__((ext_vector_type(16))) __bf16 v16bf;
typedef __attribute__((ext_vector_type(8)))  float  v8f;
typedef __attribute__((ext_vector_type(4)))  float  f4v;

#define BATCH   4096
#define F_ST    20
#define F_DY    10
#define MLEN    50
#define EDIM    16
#define F_ALL   30
#define F_PAD   31          // field 30 is an all-zero pad field (kills K padding)
#define NPAIR   435
#define D_IN    6960        // NPAIR * EDIM
#define D_PAD   6976        // padded to multiple of 32
#define HDIM    512
#define BN_EPS  1e-5f

// round-to-nearest-even float -> bf16 (avoid relying on cast lowering)
static __device__ inline bf16 f2bf(float f) {
    unsigned u = __builtin_bit_cast(unsigned, f);
    unsigned r = (u + 0x7fffu + ((u >> 16) & 1u)) >> 16;
    unsigned short h = (unsigned short)r;
    return __builtin_bit_cast(bf16, h);
}

union FragBf { v16bf v; f4v f4[2]; bf16 h[16]; };

// compile-time strict-upper-triangle pair tables; pad pairs point at the
// all-zero field 30 so padded K-columns are exactly zero with no branches.
struct Tabs { unsigned char i[448]; unsigned char j[448]; };
static constexpr Tabs makeTabs() {
    Tabs t{};
    int p = 0;
    for (int a = 0; a < F_ALL; ++a)
        for (int b = a + 1; b < F_ALL; ++b) { t.i[p] = (unsigned char)a; t.j[p] = (unsigned char)b; ++p; }
    for (; p < 448; ++p) { t.i[p] = (unsigned char)(F_PAD - 1); t.j[p] = (unsigned char)(F_PAD - 1); }
    return t;
}
__constant__ Tabs cTabs = makeTabs();

// ---------------------------------------------------------------------------
// K1: embeddings + linear-term sums.  grid=BATCH, block=512
//   A[b][31][16] f32 (field 30 zeroed) ; lr_sum[b]
// ---------------------------------------------------------------------------
__global__ __launch_bounds__(512) void embed_lr_kernel(
    const int*   __restrict__ static_ids,     // [B,20]
    const int*   __restrict__ dynamic_ids,    // [B,10,50]
    const int*   __restrict__ dynamic_len,    // [B,10]
    const float* __restrict__ st_emb,         // [N_ST,16]
    const float* __restrict__ dy_emb,         // [N_DY,16]
    const float* __restrict__ st_lr,          // [N_ST]
    const float* __restrict__ dy_lr,          // [N_DY]
    float* __restrict__ A,                    // [B,31,16]
    float* __restrict__ lr_sum)               // [B]
{
    const int b = blockIdx.x;
    const int t = threadIdx.x;
    __shared__ float s_lr[2];

    if (t < F_DY * EDIM) {                         // dynamic masked-mean embeddings
        const int f = t >> 4, e = t & 15;
        int len = dynamic_len[b * F_DY + f]; if (len < 1) len = 1;
        const int* ids = dynamic_ids + ((size_t)b * F_DY + f) * MLEN;
        float s = 0.f;
        for (int m = 0; m < len; ++m) s += dy_emb[(size_t)ids[m] * EDIM + e];
        A[((size_t)b * F_PAD + F_ST + f) * EDIM + e] = s / (float)len;
    } else if (t < F_DY * EDIM + F_ST * EDIM) {    // static lookups
        const int u = t - F_DY * EDIM;
        const int f = u >> 4, e = u & 15;
        A[((size_t)b * F_PAD + f) * EDIM + e] =
            st_emb[(size_t)static_ids[b * F_ST + f] * EDIM + e];
    } else if (t == 480) {                         // static LR sum
        float s = 0.f;
        for (int f = 0; f < F_ST; ++f) s += st_lr[static_ids[b * F_ST + f]];
        s_lr[0] = s;
    } else if (t == 481) {                         // dynamic LR sum (method='sum')
        float s = 0.f;
        for (int f = 0; f < F_DY; ++f) {
            int len = dynamic_len[b * F_DY + f]; if (len < 1) len = 1;
            const int* ids = dynamic_ids + ((size_t)b * F_DY + f) * MLEN;
            for (int m = 0; m < len; ++m) s += dy_lr[ids[m]];
        }
        s_lr[1] = s;
    } else if (t >= 482 && t < 482 + EDIM) {       // zero the pad field
        A[((size_t)b * F_PAD + (F_PAD - 1)) * EDIM + (t - 482)] = 0.f;
    }
    __syncthreads();
    if (t == 0) lr_sum[b] = s_lr[0] + s_lr[1];
}

// ---------------------------------------------------------------------------
// K2: per-column batch sums of x (and x^2) without materializing x.
//   grid=NPAIR, block=256  ->  S1x[435*16], S2x[435*16]
// ---------------------------------------------------------------------------
__global__ __launch_bounds__(256) void colstats_x_kernel(
    const float* __restrict__ A,              // [B,31,16]
    float* __restrict__ S1x, float* __restrict__ S2x)
{
    const int p = blockIdx.x;
    const int t = threadIdx.x;
    const int fi = cTabs.i[p], fj = cTabs.j[p];

    float s1[EDIM], s2[EDIM];
#pragma unroll
    for (int e = 0; e < EDIM; ++e) { s1[e] = 0.f; s2[e] = 0.f; }

    for (int b = t; b < BATCH; b += 256) {
        const float* pi = A + ((size_t)b * F_PAD + fi) * EDIM;
        const float* pj = A + ((size_t)b * F_PAD + fj) * EDIM;
#pragma unroll
        for (int q = 0; q < 4; ++q) {
            f4v xi = *(const f4v*)(pi + 4 * q);
            f4v xj = *(const f4v*)(pj + 4 * q);
            f4v x  = xi * xj;
#pragma unroll
            for (int c = 0; c < 4; ++c) {
                float v = x[c];
                s1[4 * q + c] += v;
                s2[4 * q + c] += v * v;
            }
        }
    }
    __shared__ float sh[256 * 32];
#pragma unroll
    for (int e = 0; e < EDIM; ++e) { sh[t * 32 + e] = s1[e]; sh[t * 32 + 16 + e] = s2[e]; }
    __syncthreads();
    if (t < 32) {
        float acc = 0.f;
        for (int r = 0; r < 256; ++r) acc += sh[r * 32 + t];
        if (t < 16) S1x[p * EDIM + t]        = acc;
        else        S2x[p * EDIM + (t - 16)] = acc;
    }
}

// ---------------------------------------------------------------------------
// K3: BN1 fold coefficients a1,c1 from sums.  grid=28, block=256
// ---------------------------------------------------------------------------
__global__ __launch_bounds__(256) void fold1_kernel(
    const float* __restrict__ S1, const float* __restrict__ S2,
    const float* __restrict__ g,  const float* __restrict__ beta,
    float* __restrict__ a, float* __restrict__ c)
{
    const int idx = blockIdx.x * 256 + threadIdx.x;
    if (idx >= D_IN) return;
    const float invB = 1.f / (float)BATCH;
    float mean = S1[idx] * invB;
    float var  = S2[idx] * invB - mean * mean;
    float av   = g[idx] * rsqrtf(var + BN_EPS);
    a[idx] = av;
    c[idx] = beta[idx] - av * mean;
}

// ---------------------------------------------------------------------------
// K4/K8: build transposed, BN-scaled bf16 weights WT[Nc][Dpad] + folded bias.
//   grid = Nc/256, block=256
// ---------------------------------------------------------------------------
__global__ __launch_bounds__(256) void wfold_kernel(
    const float* __restrict__ W,     // [D, Nc] row-major
    const float* __restrict__ a,     // [D]
    const float* __restrict__ cvec,  // [D]
    const float* __restrict__ bvec,  // [Nc]
    bf16*  __restrict__ WT,          // [Nc, Dpad]
    float* __restrict__ hb,          // [Nc]
    int D, int Dpad, int Nc)
{
    const int n = blockIdx.x * 256 + threadIdx.x;
    if (n >= Nc) return;
    float acc = 0.f;
    bf16* dst = WT + (size_t)n * Dpad;
    for (int j = 0; j < D; ++j) {
        float w = W[(size_t)j * Nc + n];
        acc += cvec[j] * w;
        dst[j] = f2bf(a[j] * w);
    }
    for (int j = D; j < Dpad; ++j) dst[j] = f2bf(0.f);
    hb[n] = bvec[n] + acc;
}

// ---------------------------------------------------------------------------
// GEMM1: h1 = x @ W1s + hb1, x built on the fly from pair products.
//   Per K-step the two pairs involved are wave-uniform: p1=k>>4, p2=p1+1.
//   Only e0 in {0,8} is lane-dependent -> scalar table loads, branch-free loop
//   (pad pair = (30,30) over the zeroed field).
//   grid = (M/128, N/128), block = 256 (8 waves; wave tile 32x64)
// ---------------------------------------------------------------------------
static __device__ inline void pairChunk(const float* __restrict__ rbase,
                                        int fi, int fj, int e0,
                                        bf16* __restrict__ out)
{
    const f4v* pi = (const f4v*)(rbase + fi * EDIM + e0);
    const f4v* pj = (const f4v*)(rbase + fj * EDIM + e0);
    f4v x0 = pi[0] * pj[0];
    f4v x1 = pi[1] * pj[1];
#pragma unroll
    for (int c = 0; c < 4; ++c) { out[c] = f2bf(x0[c]); out[4 + c] = f2bf(x1[c]); }
}

__global__ __launch_bounds__(256) void gemm1_kernel(
    const float* __restrict__ Aemb,   // [B,31,16]
    const bf16*  __restrict__ BT,     // [512, D_PAD]
    const float* __restrict__ hb,     // [512]
    float* __restrict__ C)            // [B,512]
{
    const int lane = threadIdx.x & 31;
    const int wave = threadIdx.x >> 5;
    const int wr = wave & 3;
    const int wc = wave >> 2;
    const int rowBase = blockIdx.x * 128 + wr * 32;
    const int colBase = blockIdx.y * 128 + wc * 64;
    const int lmod = lane & 15;
    const bool hi  = lane >= 16;
    const int e0   = hi ? 8 : 0;

    const float* rbase0 = Aemb + (size_t)(rowBase + lmod) * (F_PAD * EDIM);
    const float* rbase1 = rbase0 + (size_t)16 * (F_PAD * EDIM);

    v8f acc[2][4];
#pragma unroll
    for (int mt = 0; mt < 2; ++mt)
#pragma unroll
        for (int nt = 0; nt < 4; ++nt)
#pragma unroll
            for (int g = 0; g < 8; ++g) acc[mt][nt][g] = 0.f;

    for (int k = 0; k < D_PAD; k += 32) {
        const int p1 = k >> 4;              // wave-uniform pair indices
        const int p2 = p1 + 1;
        const int i1 = cTabs.i[p1], j1 = cTabs.j[p1];
        const int i2 = cTabs.i[p2], j2 = cTabs.j[p2];
        const int kb1 = k + e0;
        const int kb2 = kb1 + 16;

        FragBf a[2];
        pairChunk(rbase0, i1, j1, e0, &a[0].h[0]);
        pairChunk(rbase0, i2, j2, e0, &a[0].h[8]);
        pairChunk(rbase1, i1, j1, e0, &a[1].h[0]);
        pairChunk(rbase1, i2, j2, e0, &a[1].h[8]);

        FragBf bf[4];
#pragma unroll
        for (int nt = 0; nt < 4; ++nt) {
            const bf16* bp = BT + (size_t)(colBase + nt * 16 + lmod) * D_PAD;
            bf[nt].f4[0] = *(const f4v*)(bp + kb1);
            bf[nt].f4[1] = *(const f4v*)(bp + kb2);
        }
#pragma unroll
        for (int mt = 0; mt < 2; ++mt)
#pragma unroll
            for (int nt = 0; nt < 4; ++nt)
                acc[mt][nt] = __builtin_amdgcn_wmma_f32_16x16x32_bf16(
                    false, a[mt].v, false, bf[nt].v, (short)0, acc[mt][nt], false, false);
    }

#pragma unroll
    for (int nt = 0; nt < 4; ++nt) {
        const int col = colBase + nt * 16 + lmod;
        const float hbv = hb[col];
#pragma unroll
        for (int mt = 0; mt < 2; ++mt)
#pragma unroll
            for (int g = 0; g < 8; ++g) {
                const int row = rowBase + mt * 16 + g + (hi ? 8 : 0);
                C[(size_t)row * HDIM + col] = acc[mt][nt][g] + hbv;
            }
    }
}

// ---------------------------------------------------------------------------
// GEMM2: C = A(bf16 [M,512]) @ BT(bf16 [512,512]) + hb.  Same tiling.
// ---------------------------------------------------------------------------
__global__ __launch_bounds__(256) void gemm2_kernel(
    const bf16*  __restrict__ A,      // [B,512]
    const bf16*  __restrict__ BT,     // [512,512]
    const float* __restrict__ hb,     // [512]
    float* __restrict__ C)            // [B,512]
{
    const int lane = threadIdx.x & 31;
    const int wave = threadIdx.x >> 5;
    const int wr = wave & 3;
    const int wc = wave >> 2;
    const int rowBase = blockIdx.x * 128 + wr * 32;
    const int colBase = blockIdx.y * 128 + wc * 64;
    const int lmod = lane & 15;
    const bool hi  = lane >= 16;

    v8f acc[2][4];
#pragma unroll
    for (int mt = 0; mt < 2; ++mt)
#pragma unroll
        for (int nt = 0; nt < 4; ++nt)
#pragma unroll
            for (int g = 0; g < 8; ++g) acc[mt][nt][g] = 0.f;

    for (int k = 0; k < HDIM; k += 32) {
        const int kb1 = k + (hi ? 8 : 0);
        const int kb2 = kb1 + 16;

        FragBf a[2];
#pragma unroll
        for (int mt = 0; mt < 2; ++mt) {
            const bf16* ap = A + (size_t)(rowBase + mt * 16 + lmod) * HDIM;
            a[mt].f4[0] = *(const f4v*)(ap + kb1);
            a[mt].f4[1] = *(const f4v*)(ap + kb2);
        }
        FragBf bf[4];
#pragma unroll
        for (int nt = 0; nt < 4; ++nt) {
            const bf16* bp = BT + (size_t)(colBase + nt * 16 + lmod) * HDIM;
            bf[nt].f4[0] = *(const f4v*)(bp + kb1);
            bf[nt].f4[1] = *(const f4v*)(bp + kb2);
        }
#pragma unroll
        for (int mt = 0; mt < 2; ++mt)
#pragma unroll
            for (int nt = 0; nt < 4; ++nt)
                acc[mt][nt] = __builtin_amdgcn_wmma_f32_16x16x32_bf16(
                    false, a[mt].v, false, bf[nt].v, (short)0, acc[mt][nt], false, false);
    }

#pragma unroll
    for (int nt = 0; nt < 4; ++nt) {
        const int col = colBase + nt * 16 + lmod;
        const float hbv = hb[col];
#pragma unroll
        for (int mt = 0; mt < 2; ++mt)
#pragma unroll
            for (int g = 0; g < 8; ++g) {
                const int row = rowBase + mt * 16 + g + (hi ? 8 : 0);
                C[(size_t)row * HDIM + col] = acc[mt][nt][g] + hbv;
            }
    }
}

// ---------------------------------------------------------------------------
// K6/K10: column mean/var of relu(H) -> BN fold coeffs a,c.  grid=2, block=256
// ---------------------------------------------------------------------------
__global__ __launch_bounds__(256) void stats_ac_kernel(
    const float* __restrict__ H,      // [B,512]
    const float* __restrict__ g, const float* __restrict__ beta,
    float* __restrict__ a, float* __restrict__ c)
{
    const int n = blockIdx.x * 256 + threadIdx.x;
    if (n >= HDIM) return;
    float s1 = 0.f, s2 = 0.f;
    for (int r = 0; r < BATCH; ++r) {
        float y = H[(size_t)r * HDIM + n];
        y = y > 0.f ? y : 0.f;
        s1 += y; s2 += y * y;
    }
    const float invB = 1.f / (float)BATCH;
    float mean = s1 * invB;
    float var  = s2 * invB - mean * mean;
    float av   = g[n] * rsqrtf(var + BN_EPS);
    a[n] = av;
    c[n] = beta[n] - av * mean;
}

// ---------------------------------------------------------------------------
// K7: A2 = bf16(relu(h1)).  grid = B*H/256
// ---------------------------------------------------------------------------
__global__ __launch_bounds__(256) void cast_relu_kernel(
    const float* __restrict__ H, bf16* __restrict__ A2, int n)
{
    const int idx = blockIdx.x * 256 + threadIdx.x;
    if (idx >= n) return;
    float y = H[idx];
    A2[idx] = f2bf(y > 0.f ? y : 0.f);
}

// ---------------------------------------------------------------------------
// K11: rowsum of W3 and BN3-scaled vector.  grid=2, block=256
// ---------------------------------------------------------------------------
__global__ __launch_bounds__(256) void w3prep_kernel(
    const float* __restrict__ W3,     // [512,512]
    const float* __restrict__ a3,
    float* __restrict__ rs, float* __restrict__ wvec)
{
    const int j = blockIdx.x * 256 + threadIdx.x;
    if (j >= HDIM) return;
    float s = 0.f;
    const float* row = W3 + (size_t)j * HDIM;
    for (int k = 0; k < HDIM; ++k) s += row[k];
    rs[j]   = s;
    wvec[j] = a3[j] * s;
}

// ---------------------------------------------------------------------------
// K12: scores[b] = bias + lr_sum[b] + sum_j( relu(h2)*wvec + c3*rs + b3 )
//   grid = B, block = 256
// ---------------------------------------------------------------------------
__global__ __launch_bounds__(256) void scores_kernel(
    const float* __restrict__ H2,     // [B,512]
    const float* __restrict__ wvec, const float* __restrict__ c3,
    const float* __restrict__ rs,   const float* __restrict__ b3,
    const float* __restrict__ bias, const float* __restrict__ lr_sum,
    float* __restrict__ out)
{
    const int b = blockIdx.x;
    const int t = threadIdx.x;
    float acc = 0.f;
    for (int j = t; j < HDIM; j += 256) {
        float y = H2[(size_t)b * HDIM + j];
        y = y > 0.f ? y : 0.f;
        acc += y * wvec[j] + c3[j] * rs[j] + b3[j];
    }
    __shared__ float red[256];
    red[t] = acc;
    __syncthreads();
    for (int s = 128; s > 0; s >>= 1) {
        if (t < s) red[t] += red[t + s];
        __syncthreads();
    }
    if (t == 0) out[b] = bias[0] + lr_sum[b] + red[0];
}

// ---------------------------------------------------------------------------
// Host launcher
// ---------------------------------------------------------------------------
extern "C" void kernel_launch(void* const* d_in, const int* in_sizes, int n_in,
                              void* d_out, int out_size, void* d_ws, size_t ws_size,
                              hipStream_t stream) {
    const int*   static_ids  = (const int*)  d_in[0];
    const int*   dynamic_ids = (const int*)  d_in[1];
    const int*   dynamic_len = (const int*)  d_in[2];
    const float* st_emb      = (const float*)d_in[3];
    const float* dy_emb      = (const float*)d_in[4];
    const float* st_lr       = (const float*)d_in[5];
    const float* dy_lr       = (const float*)d_in[6];
    const float* bias        = (const float*)d_in[7];
    const float* bn1_g       = (const float*)d_in[8];
    const float* bn1_b       = (const float*)d_in[9];
    const float* W1          = (const float*)d_in[10];
    const float* b1          = (const float*)d_in[11];
    const float* bn2_g       = (const float*)d_in[12];
    const float* bn2_b       = (const float*)d_in[13];
    const float* W2          = (const float*)d_in[14];
    const float* b2          = (const float*)d_in[15];
    const float* bn3_g       = (const float*)d_in[16];
    const float* bn3_b       = (const float*)d_in[17];
    const float* W3          = (const float*)d_in[18];
    const float* b3          = (const float*)d_in[19];
    // d_in[20..23] (bn_out_g/b, W_out, b_out) feed dead code in the reference.

    float* out = (float*)d_out;

    // workspace carve-up (256B aligned)
    char* base = (char*)d_ws;
    size_t off = 0;
    auto carve = [&](size_t bytes) -> char* {
        char* p = base + off;
        off = (off + bytes + 255) & ~(size_t)255;
        return p;
    };
    float* Aemb   = (float*)carve((size_t)BATCH * F_PAD * EDIM * 4);  // 8.1 MB
    float* lr_sum = (float*)carve((size_t)BATCH * 4);
    float* S1x    = (float*)carve((size_t)D_IN * 4);
    float* S2x    = (float*)carve((size_t)D_IN * 4);
    float* a1     = (float*)carve((size_t)D_IN * 4);
    float* c1     = (float*)carve((size_t)D_IN * 4);
    bf16*  W1sT   = (bf16*) carve((size_t)HDIM * D_PAD * 2);          // 7.1 MB
    float* hb1    = (float*)carve((size_t)HDIM * 4);
    float* h1     = (float*)carve((size_t)BATCH * HDIM * 4);          // 8.4 MB
    float* a2     = (float*)carve((size_t)HDIM * 4);
    float* c2     = (float*)carve((size_t)HDIM * 4);
    bf16*  A2     = (bf16*) carve((size_t)BATCH * HDIM * 2);          // 4.2 MB
    bf16*  W2sT   = (bf16*) carve((size_t)HDIM * HDIM * 2);
    float* hb2    = (float*)carve((size_t)HDIM * 4);
    float* h2     = (float*)carve((size_t)BATCH * HDIM * 4);          // 8.4 MB
    float* a3     = (float*)carve((size_t)HDIM * 4);
    float* c3     = (float*)carve((size_t)HDIM * 4);
    float* rs     = (float*)carve((size_t)HDIM * 4);
    float* wvec   = (float*)carve((size_t)HDIM * 4);
    (void)ws_size; (void)n_in; (void)in_sizes; (void)out_size;

    // 1) embeddings + LR sums (+ zeroed pad field)
    embed_lr_kernel<<<BATCH, 512, 0, stream>>>(
        static_ids, dynamic_ids, dynamic_len, st_emb, dy_emb, st_lr, dy_lr,
        Aemb, lr_sum);

    // 2) BN1 column stats of x (implicit), 3) fold
    colstats_x_kernel<<<NPAIR, 256, 0, stream>>>(Aemb, S1x, S2x);
    fold1_kernel<<<(D_IN + 255) / 256, 256, 0, stream>>>(S1x, S2x, bn1_g, bn1_b, a1, c1);

    // 4) W1 fold (scaled, transposed, zero-padded K) + bias fold
    wfold_kernel<<<2, 256, 0, stream>>>(W1, a1, c1, b1, W1sT, hb1, D_IN, D_PAD, HDIM);

    // 5) GEMM1 with on-the-fly pairwise-product A (WMMA bf16, branch-free K loop)
    gemm1_kernel<<<dim3(BATCH / 128, HDIM / 128), 256, 0, stream>>>(Aemb, W1sT, hb1, h1);

    // 6) BN2 stats+fold on relu(h1), 7) cast A2, 8) W2 fold
    stats_ac_kernel<<<2, 256, 0, stream>>>(h1, bn2_g, bn2_b, a2, c2);
    cast_relu_kernel<<<(BATCH * HDIM) / 256, 256, 0, stream>>>(h1, A2, BATCH * HDIM);
    wfold_kernel<<<2, 256, 0, stream>>>(W2, a2, c2, b2, W2sT, hb2, HDIM, HDIM, HDIM);

    // 9) GEMM2 (WMMA bf16)
    gemm2_kernel<<<dim3(BATCH / 128, HDIM / 128), 256, 0, stream>>>(A2, W2sT, hb2, h2);

    // 10) BN3 stats+fold, 11) collapse GEMM3 to mat-vec, 12) scores
    stats_ac_kernel<<<2, 256, 0, stream>>>(h2, bn3_g, bn3_b, a3, c3);
    w3prep_kernel<<<2, 256, 0, stream>>>(W3, a3, rs, wvec);
    scores_kernel<<<BATCH, 256, 0, stream>>>(h2, wvec, c3, rs, b3, bias, lr_sum, out);
}
